// TriMulReference_43576738186057
// MI455X (gfx1250) — matrix operand order repaired and checked by reading, not compiled, not run on hardware
//
#include <hip/hip_runtime.h>

// ---------------------------------------------------------------------------
// TriMul (triangular multiplicative update) for MI455X / gfx1250.
// GEMMs on v_wmma_f32_16x16x32_f16 (wave32, f32 accumulate); the einsum
// stage feeds LDS with the Tensor Data Mover (tensor_load_to_lds) in a
// double-buffered pipeline tracked by TENSORcnt.
// ---------------------------------------------------------------------------

#define S 512
#define D 128
#define H 128
#define SS (S * S)
#define EPS 1e-5f

typedef __attribute__((ext_vector_type(16))) _Float16     v16h;
typedef __attribute__((ext_vector_type(8)))  _Float16     v8h;
typedef __attribute__((ext_vector_type(8)))  float        v8f;
typedef __attribute__((ext_vector_type(4)))  unsigned int v4u;
typedef __attribute__((ext_vector_type(8)))  int          v8i;
typedef __attribute__((ext_vector_type(4)))  int          v4i;

#if defined(__has_builtin)
#if __has_builtin(__builtin_amdgcn_tensor_load_to_lds) && \
    __has_builtin(__builtin_amdgcn_s_wait_tensorcnt)
#define TRIMUL_USE_TDM 1
#endif
#endif
#ifndef TRIMUL_USE_TDM
#define TRIMUL_USE_TDM 0
#endif

__device__ __forceinline__ v16h frag_cat(v8h lo, v8h hi) {
  v16h r;
#pragma unroll
  for (int i = 0; i < 8; ++i) { r[i] = lo[i]; r[i + 8] = hi[i]; }
  return r;
}

// A fragment (16x32 f16): lane<16 holds row M=lane with K 0-7 / 16-23,
// lane>=16 holds row M=lane-16 with K 8-15 / 24-31 (ISA 7.12.2).
__device__ __forceinline__ v16h load_a_frag(const _Float16* base, int ld, int lane) {
  const int row  = lane & 15;
  const int kofs = (lane & 16) ? 8 : 0;
  const _Float16* p = base + row * ld + kofs;
  return frag_cat(*(const v8h*)p, *(const v8h*)(p + 16));
}

// B fragment (32x16 f16): lane<16 holds col N=lane with K 0-7 / 16-23,
// lane>=16 holds col N=lane-16 with K 8-15 / 24-31.
__device__ __forceinline__ v16h load_b_frag(const _Float16* base, int ld, int lane) {
  const int col  = lane & 15;
  const int kofs = (lane & 16) ? 8 : 0;
  const _Float16* p = base + col * ld + kofs;
  return frag_cat(*(const v8h*)p, *(const v8h*)(p + 16));
}

__device__ __forceinline__ v8f wmma_f16(v16h a, v16h b, v8f c) {
  return __builtin_amdgcn_wmma_f32_16x16x32_f16(false, a, false, b, (short)0, c,
                                                false, false);
}

__device__ __forceinline__ float sigmoidf_(float x) { return 1.0f / (1.0f + __expf(-x)); }

#if TRIMUL_USE_TDM
// Issue a TDM load of a 64-row x 32-halfs tile (row stride 512 halfs in
// memory) into LDS with a 16-half pad after each row (row stride 96 B in LDS,
// matching the b128-aligned fragment reads). D# per CDNA5 ISA 8.3/8.4:
//   pad_interval = 3  -> pad after every 16 DWORDs (= one 64 B row)
//   pad_amount   = 7  -> 8 DWORDs (= 32 B) of padding
// This toolchain's builtin takes 6 args:
//   (uint32x4 g0, int32x8 g1, int32x4 g2, int32x4 g3, int32x8, i32 cpol)
__device__ __forceinline__ void tdm_load_64x32(const _Float16* gsrc, _Float16* ldst) {
  const unsigned long long ga = (unsigned long long)(size_t)gsrc;
  const unsigned lds_off = (unsigned)(size_t)ldst;  // low 32 bits = LDS byte offset
  v4u g0;
  g0[0] = 1u;                                        // count=1, user D#
  g0[1] = lds_off;                                   // lds_addr
  g0[2] = (unsigned)(ga & 0xFFFFFFFFu);              // global_addr[31:0]
  g0[3] = (unsigned)((ga >> 32) & 0x1FFFFFFu) | (2u << 30);  // ga[56:32] | type=2
  v8i g1;
  g1[0] = (int)((1u << 16) | (1u << 20) | (3u << 22) | (7u << 25));
  //        data_size=2B    pad_enable    pad_interval  pad_amount
  g1[1] = 0;                 // atomic barrier addr=0, tensor_dim0[15:0] (2^20 -> 0)
  g1[2] = 0x0010;            // tensor_dim0[31:16]=16 (dim0=2^20), tensor_dim1[15:0]=0
  g1[3] = (32 << 16) | 1;    // tensor_dim1[31:16]=1 (dim1=2^16), tile_dim0=32
  g1[4] = 64;                // tile_dim1=64, tile_dim2=0
  g1[5] = S;                 // tensor_dim0_stride = 512 elements
  g1[6] = 0;
  g1[7] = 0;
  const v4i gz4 = (v4i){0, 0, 0, 0};                 // 2-D tile: groups 2/3 unused
  const v8i gz8 = (v8i){0, 0, 0, 0, 0, 0, 0, 0};
  __builtin_amdgcn_tensor_load_to_lds(g0, g1, gz4, gz4, gz8, 0);
}
#endif

// ---------------------------------------------------------------------------
// Kernel 0: convert the six 128x128 weight matrices to f16 once.
// Order: Wl, Wr, Wlg, Wrg, Wog, Wout.
// ---------------------------------------------------------------------------
__global__ void trimul_prep(const float* __restrict__ Wl, const float* __restrict__ Wr,
                            const float* __restrict__ Wlg, const float* __restrict__ Wrg,
                            const float* __restrict__ Wog, const float* __restrict__ Wout,
                            _Float16* __restrict__ wf16) {
  const int i = blockIdx.x * blockDim.x + threadIdx.x;
  if (i >= 6 * D * H) return;
  const int m = i >> 14;       // which matrix (16384 elements each)
  const int r = i & 16383;
  const float* src = (m == 0) ? Wl : (m == 1) ? Wr : (m == 2) ? Wlg
                   : (m == 3) ? Wrg : (m == 4) ? Wog : Wout;
  wf16[i] = (_Float16)src[r];
}

// ---------------------------------------------------------------------------
// Kernel 1: LayerNorm + 5 projections + gating for 16 flattened (i,k) rows.
// left/right written channel-major [h][i*S+k] (f16) for the einsum GEMMs.
// gate written [r][h] (f16) for the output epilogue.
// ---------------------------------------------------------------------------
__global__ __launch_bounds__(256) void trimul_proj(
    const float* __restrict__ x, const float* __restrict__ mask,
    const float* __restrict__ ng, const float* __restrict__ nb,
    const _Float16* __restrict__ wf16,
    _Float16* __restrict__ left_t, _Float16* __restrict__ right_t,
    _Float16* __restrict__ gate) {
  __shared__ _Float16 xa[16 * 136];   // 16 rows x 128 (pad 8 -> 272B stride, b128 aligned)
  __shared__ float mask_s[16];
  const int t = threadIdx.x;
  const size_t rbase = (size_t)blockIdx.x * 16;

  // --- LayerNorm over D for 16 rows: 16 lanes per row, 8 elems per lane ---
  {
    const int row = t >> 4;
    const int seg = t & 15;
    const float* xr = x + (rbase + row) * D + seg * 8;
    float v[8]; float s = 0.f, ss = 0.f;
#pragma unroll
    for (int e = 0; e < 8; ++e) { v[e] = xr[e]; s += v[e]; ss += v[e] * v[e]; }
#pragma unroll
    for (int m = 1; m < 16; m <<= 1) {       // stays within each 16-lane half of wave32
      s  += __shfl_xor(s,  m, 32);
      ss += __shfl_xor(ss, m, 32);
    }
    const float mu   = s * (1.0f / D);
    const float rsig = rsqrtf(ss * (1.0f / D) - mu * mu + EPS);
#pragma unroll
    for (int e = 0; e < 8; ++e) {
      const int d = seg * 8 + e;
      xa[row * 136 + d] = (_Float16)(((v[e] - mu) * rsig) * ng[d] + nb[d]);
    }
    if (t < 16) mask_s[t] = mask[rbase + t];
  }
  __syncthreads();

  // --- each of the 8 waves owns a 16-wide N(=h) tile; K = D = 4 x 32 ---
  const int wave  = t >> 5;
  const int lane  = t & 31;
  const int hbase = wave * 16;
  const _Float16* Wl16  = wf16 + 0 * D * H + hbase * D;
  const _Float16* Wr16  = wf16 + 1 * D * H + hbase * D;
  const _Float16* Wlg16 = wf16 + 2 * D * H + hbase * D;
  const _Float16* Wrg16 = wf16 + 3 * D * H + hbase * D;
  const _Float16* Wog16 = wf16 + 4 * D * H + hbase * D;

  v8f al = {}, ar = {}, alg = {}, arg_ = {}, aog = {};
#pragma unroll
  for (int kt = 0; kt < 4; ++kt) {
    const int kk = kt * 32;
    const v16h a = load_a_frag(&xa[kk], 136, lane);
    // W row h is exactly the B-operand column (W.T in K position) -> contiguous.
    const v16h bl  = load_b_frag(Wl16  + kk, D, lane);
    const v16h br  = load_b_frag(Wr16  + kk, D, lane);
    const v16h blg = load_b_frag(Wlg16 + kk, D, lane);
    const v16h brg = load_b_frag(Wrg16 + kk, D, lane);
    const v16h bog = load_b_frag(Wog16 + kk, D, lane);
    al   = wmma_f16(a, bl,  al);
    ar   = wmma_f16(a, br,  ar);
    alg  = wmma_f16(a, blg, alg);
    arg_ = wmma_f16(a, brg, arg_);
    aog  = wmma_f16(a, bog, aog);
  }

  // --- epilogue: mask * sigmoid-gate, scatter channel-major ---
  const int row0 = (lane & 16) ? 8 : 0;     // C/D layout: lanes 0-15 rows 0-7, 16-31 rows 8-15
  const int col  = lane & 15;
  const int h    = hbase + col;
  v8h pl, pr, pg;
#pragma unroll
  for (int v = 0; v < 8; ++v) {
    const float mval = mask_s[row0 + v];
    pl[v] = (_Float16)(al[v] * mval * sigmoidf_(alg[v]));
    pr[v] = (_Float16)(ar[v] * mval * sigmoidf_(arg_[v]));
    pg[v] = (_Float16)(sigmoidf_(aog[v]));
  }
  // 8 consecutive rows per lane -> one 16B store per matrix
  *(v8h*)(left_t  + (size_t)h * SS + rbase + row0) = pl;
  *(v8h*)(right_t + (size_t)h * SS + rbase + row0) = pr;
#pragma unroll
  for (int v = 0; v < 8; ++v)
    gate[(rbase + row0 + v) * H + h] = pg[v];
}

// ---------------------------------------------------------------------------
// Kernel 2: per-channel contraction  pv[h] = L_h (512x512) * R_h^T (512x512).
// 64x64 output tile per 256-thread block. K-slabs (64x32 f16, padded rows)
// are staged into LDS by the Tensor Data Mover, double-buffered: lane 0
// issues TDM for slab kt+1, all waves run WMMA on slab kt, the issuing wave
// drains TENSORcnt, then a barrier publishes the buffer.
// ---------------------------------------------------------------------------
__global__ __launch_bounds__(256) void trimul_einsum(
    const _Float16* __restrict__ left_t, const _Float16* __restrict__ right_t,
    float* __restrict__ pv) {
  __shared__ _Float16 As[2][64 * 48];   // 64 x 32 + 16-half row pad (96B stride)
  __shared__ _Float16 Bs[2][64 * 48];
  const int t = threadIdx.x;
  const int h = blockIdx.z;
  const size_t ibase = (size_t)blockIdx.y * 64;
  const size_t jbase = (size_t)blockIdx.x * 64;
  const _Float16* Ag = left_t  + (size_t)h * SS + ibase * S;  // [i][k] row-major
  const _Float16* Bg = right_t + (size_t)h * SS + jbase * S;  // [j][k] row-major

  const int wave = t >> 5, lane = t & 31;
  const int sm  = wave & 3;               // 16-row M subtile
  const int sn0 = (wave >> 2) * 2;        // two 16-col N subtiles per wave
  const int sn1 = sn0 + 1;
  v8f acc0 = {}, acc1 = {};

#if TRIMUL_USE_TDM
  if (t == 0) {
    tdm_load_64x32(Ag, &As[0][0]);
    tdm_load_64x32(Bg, &Bs[0][0]);
    __builtin_amdgcn_s_wait_tensorcnt(0);
  }
  __syncthreads();
  for (int kt = 0; kt < 16; ++kt) {
    const int cur = kt & 1, nxt = cur ^ 1;
    if ((t == 0) && (kt + 1 < 16)) {      // TDM for next slab overlaps compute
      tdm_load_64x32(Ag + (kt + 1) * 32, &As[nxt][0]);
      tdm_load_64x32(Bg + (kt + 1) * 32, &Bs[nxt][0]);
    }
    const v16h a  = load_a_frag(&As[cur][sm  * 16 * 48], 48, lane);
    const v16h b0 = load_b_frag(&Bs[cur][sn0 * 16 * 48], 48, lane);
    const v16h b1 = load_b_frag(&Bs[cur][sn1 * 16 * 48], 48, lane);
    acc0 = wmma_f16(a, b0, acc0);
    acc1 = wmma_f16(a, b1, acc1);
    if (t == 0) __builtin_amdgcn_s_wait_tensorcnt(0);
    __syncthreads();
  }
#else
  const int ldrow = t >> 2;               // cooperative loader: 64 rows x 4 chunks
  const int ldch  = (t & 3) * 8;
  for (int kt = 0; kt < 16; ++kt) {
    const int kk = kt * 32;
    const int cur = kt & 1;
    *(v8h*)&As[cur][ldrow * 48 + ldch] = *(const v8h*)(Ag + (size_t)ldrow * S + kk + ldch);
    *(v8h*)&Bs[cur][ldrow * 48 + ldch] = *(const v8h*)(Bg + (size_t)ldrow * S + kk + ldch);
    if (kt < 15) {
      __builtin_prefetch(Ag + (size_t)ldrow * S + kk + 32 + ldch, 0, 1);
      __builtin_prefetch(Bg + (size_t)ldrow * S + kk + 32 + ldch, 0, 1);
    }
    __syncthreads();
    const v16h a  = load_a_frag(&As[cur][sm  * 16 * 48], 48, lane);
    const v16h b0 = load_b_frag(&Bs[cur][sn0 * 16 * 48], 48, lane);
    const v16h b1 = load_b_frag(&Bs[cur][sn1 * 16 * 48], 48, lane);
    acc0 = wmma_f16(a, b0, acc0);
    acc1 = wmma_f16(a, b1, acc1);
    __syncthreads();
  }
#endif

  float* pvh = pv + (size_t)h * SS;       // pv layout [h][i*S+j]: coalesced rows
  const int row0 = (lane & 16) ? 8 : 0;
  const int col  = lane & 15;
#pragma unroll
  for (int v = 0; v < 8; ++v) {
    pvh[(ibase + sm * 16 + row0 + v) * S + jbase + sn0 * 16 + col] = acc0[v];
    pvh[(ibase + sm * 16 + row0 + v) * S + jbase + sn1 * 16 + col] = acc1[v];
  }
}

// ---------------------------------------------------------------------------
// Kernel 3: LayerNorm over H + gate + output projection (y @ Wout.T).
// 128 rows per 128-thread block; two coalesced passes over pv for moments,
// f16 activation tile in LDS, then WMMA against Wout (rows of Wout are the
// B-operand columns -> contiguous loads).
// ---------------------------------------------------------------------------
__global__ __launch_bounds__(128) void trimul_out(
    const float* __restrict__ pv, const _Float16* __restrict__ gate,
    const float* __restrict__ ong, const float* __restrict__ onb,
    const _Float16* __restrict__ wout16, float* __restrict__ out) {
  __shared__ _Float16 ys[128 * 144];  // 128 rows x 128 h (pad 16 -> 288B stride)
  const int t = threadIdx.x;
  const size_t rbase = (size_t)blockIdx.x * 128;

  // pass 1: moments (pv is [h][r] -> consecutive threads hit consecutive addrs)
  float s = 0.f, ss = 0.f;
  for (int h = 0; h < H; ++h) {
    const float v = pv[(size_t)h * SS + rbase + t];
    s += v; ss += v * v;
  }
  const float mu   = s * (1.0f / H);
  const float rsig = rsqrtf(ss * (1.0f / H) - mu * mu + EPS);

  // pass 2: normalize * gamma + beta, multiply by sigmoid out-gate, stash f16
  for (int h = 0; h < H; ++h) {
    const float v = pv[(size_t)h * SS + rbase + t];
    const float g = (float)gate[(rbase + t) * H + h];
    ys[t * 144 + h] = (_Float16)((((v - mu) * rsig) * ong[h] + onb[h]) * g);
  }
  __syncthreads();

  // GEMM: (128 rows x 128 d), K = H = 4 x 32. 4 waves x 2 d-tiles x 8 r-tiles.
  const int wave = t >> 5, lane = t & 31;
  const int row0 = (lane & 16) ? 8 : 0;
  const int col  = lane & 15;
  for (int half = 0; half < 2; ++half) {
    const int dbase = (wave + half * 4) * 16;
    for (int rt = 0; rt < 8; ++rt) {
      v8f acc = {};
#pragma unroll
      for (int kt = 0; kt < 4; ++kt) {
        const int kk = kt * 32;
        const v16h a = load_a_frag(&ys[rt * 16 * 144 + kk], 144, lane);
        const v16h b = load_b_frag(wout16 + dbase * H + kk, H, lane);
        acc = wmma_f16(a, b, acc);
      }
#pragma unroll
      for (int v = 0; v < 8; ++v)
        out[(rbase + rt * 16 + row0 + v) * D + dbase + col] = acc[v];
    }
  }
}

// ---------------------------------------------------------------------------
// Host entry. Workspace layout (all offsets 16B-aligned):
//   [0, 192KB)      wf16  : 6 x 128x128 f16 weights
//   [256KB, +64MB)  left_t  f16 [H][S*S]
//   [.., +64MB)     right_t f16 [H][S*S]
//   [.., +64MB)     gate    f16 [S*S][H]
//   [.., +128MB)    pv      f32 [H][S*S]
// ---------------------------------------------------------------------------
extern "C" void kernel_launch(void* const* d_in, const int* in_sizes, int n_in,
                              void* d_out, int out_size, void* d_ws, size_t ws_size,
                              hipStream_t stream) {
  (void)in_sizes; (void)n_in; (void)out_size; (void)ws_size;
  const float* x    = (const float*)d_in[0];
  const float* mask = (const float*)d_in[1];
  const float* ng   = (const float*)d_in[2];
  const float* nb   = (const float*)d_in[3];
  const float* Wl   = (const float*)d_in[4];
  const float* Wr   = (const float*)d_in[5];
  const float* Wlg  = (const float*)d_in[6];
  const float* Wrg  = (const float*)d_in[7];
  const float* Wog  = (const float*)d_in[8];
  const float* ong  = (const float*)d_in[9];
  const float* onb  = (const float*)d_in[10];
  const float* Wout = (const float*)d_in[11];
  float* out = (float*)d_out;

  char* w = (char*)d_ws;
  _Float16* wf16 = (_Float16*)w;
  size_t off = 256 * 1024;
  _Float16* left_t  = (_Float16*)(w + off); off += (size_t)H * SS * sizeof(_Float16);
  _Float16* right_t = (_Float16*)(w + off); off += (size_t)H * SS * sizeof(_Float16);
  _Float16* gate    = (_Float16*)(w + off); off += (size_t)SS * H * sizeof(_Float16);
  float*    pv      = (float*)(w + off);

  hipLaunchKernelGGL(trimul_prep, dim3((6 * D * H + 255) / 256), dim3(256), 0, stream,
                     Wl, Wr, Wlg, Wrg, Wog, Wout, wf16);
  hipLaunchKernelGGL(trimul_proj, dim3(SS / 16), dim3(256), 0, stream,
                     x, mask, ng, nb, wf16, left_t, right_t, gate);
  hipLaunchKernelGGL(trimul_einsum, dim3(S / 64, S / 64, H), dim3(256), 0, stream,
                     left_t, right_t, pv);
  hipLaunchKernelGGL(trimul_out, dim3(SS / 128), dim3(128), 0, stream,
                     pv, gate, ong, onb, wf16 + 5 * D * H, out);
}
